// MLP_56341380989394
// MI455X (gfx1250) — matrix operand order repaired and checked by reading, not compiled, not run on hardware
//
#include <hip/hip_runtime.h>
#include <hip/hip_bf16.h>

// ---------------------------------------------------------------------------
// MLP: out = (relu(relu(relu(x W0^T + b0) W1^T + b1) W2^T + b2)) W3^T + b3
// Pass 1: convert all weights f32 -> f16 (one-time, ~12us of HBM traffic).
// Pass 2: four NT GEMMs, f16 operands, f32 accumulate via
//         V_WMMA_F32_16X16X32_F16, tiles DMA'd into LDS with
//         GLOBAL_LOAD_ASYNC_TO_LDS_B128 (ASYNCcnt), double-buffered.
// ---------------------------------------------------------------------------

typedef __attribute__((ext_vector_type(16))) _Float16 v16h;
typedef __attribute__((ext_vector_type(8)))  _Float16 v8h;
typedef __attribute__((ext_vector_type(8)))  float    v8f;

#define BM 128
#define BN 128
#define BK 64
#define LDS_STRIDE 72   // 64 + 8 halves pad (144B rows): conflict-free b128 frag reads

// ---- one-time f32 -> f16 weight conversion (n % 8 == 0) -------------------
__global__ __launch_bounds__(256) void cvt_f32_to_f16(const float* __restrict__ src,
                                                      _Float16* __restrict__ dst,
                                                      int n8)
{
    int i = blockIdx.x * blockDim.x + threadIdx.x;
    const int stride = gridDim.x * blockDim.x;
    for (; i < n8; i += stride) {
        const float4* p = reinterpret_cast<const float4*>(src + (size_t)i * 8);
        float4 f0 = p[0], f1 = p[1];
        v8h h = { (_Float16)f0.x,(_Float16)f0.y,(_Float16)f0.z,(_Float16)f0.w,
                  (_Float16)f1.x,(_Float16)f1.y,(_Float16)f1.z,(_Float16)f1.w };
        *reinterpret_cast<v8h*>(dst + (size_t)i * 8) = h;
    }
}

// ---- async DMA: 64 contiguous bytes global -> LDS (ASYNCcnt, 4 ops) -------
__device__ __forceinline__ void async_copy64(const _Float16* gsrc, _Float16* ldst)
{
    uint32_t l = (uint32_t)(uintptr_t)(void*)ldst;      // LDS byte offset (low 32 bits)
    uint64_t g = (uint64_t)(uintptr_t)gsrc;
    asm volatile(
        "global_load_async_to_lds_b128 %0, %1, off\n\t"
        "global_load_async_to_lds_b128 %0, %1, off offset:16\n\t"
        "global_load_async_to_lds_b128 %0, %1, off offset:32\n\t"
        "global_load_async_to_lds_b128 %0, %1, off offset:48"
        :: "v"(l), "v"(g) : "memory");
}

// ---------------------------------------------------------------------------
// GEMM: C[M,N] = act(A[M,K] * Wh[N,K]^T + b).  AT = float (layer 0) or _Float16.
// ---------------------------------------------------------------------------
template <typename AT, typename OT, bool RELU>
__global__ __launch_bounds__(256) void
gemm_bias_act(const AT* __restrict__ A,
              const _Float16* __restrict__ Wh,
              const float* __restrict__ bias,
              OT* __restrict__ C,
              int M, int N, int K)
{
    __shared__ _Float16 sA[2][BM * LDS_STRIDE];
    __shared__ _Float16 sB[2][BN * LDS_STRIDE];

    const int tid  = threadIdx.x;
    const int lane = tid & 31;
    const int wave = tid >> 5;          // 0..7
    const int waveM = wave & 3;         // 32 rows each
    const int waveN = wave >> 2;        // 64 cols each

    const int blockRow = blockIdx.y * BM;
    const int blockCol = blockIdx.x * BN;

    // Staging: 2 threads per tile row, 32 halves (64B) each.
    const int srow = tid >> 1;          // 0..127
    const int scol = (tid & 1) * 32;    // 0 or 32 within BK

    const int laneRow = lane & 15;
    const bool hiHalf = lane >= 16;
    const int aKlo = hiHalf ? 8 : 0;    // ISA 16-bit A layout (16x32 per ks slice)
    const int aKhi = hiHalf ? 24 : 16;
    const int bKoff = hiHalf ? 16 : 0;  // unused; B frags are read with aKlo/aKhi, same k order as A

    v8f acc[2][4] = {};

    const int nkt = K / BK;

    // ---- tile stagers ---------------------------------------------------
    auto stageA = [&](int buf, int k0) {
        if constexpr (sizeof(AT) == 4) {    // layer 0: f32 -> f16 through VGPRs
            const float* g = A + (size_t)(blockRow + srow) * K + (k0 + scol);
            _Float16* dst = &sA[buf][srow * LDS_STRIDE + scol];
            const float4* p = reinterpret_cast<const float4*>(g);
#pragma unroll
            for (int c = 0; c < 4; ++c) {
                float4 f0 = p[2 * c], f1 = p[2 * c + 1];
                v8h h = { (_Float16)f0.x,(_Float16)f0.y,(_Float16)f0.z,(_Float16)f0.w,
                          (_Float16)f1.x,(_Float16)f1.y,(_Float16)f1.z,(_Float16)f1.w };
                *reinterpret_cast<v8h*>(dst + 8 * c) = h;
            }
        } else {                            // f16 activations: DMA straight to LDS
            const _Float16* g = (const _Float16*)A + (size_t)(blockRow + srow) * K + (k0 + scol);
            async_copy64(g, &sA[buf][srow * LDS_STRIDE + scol]);
        }
    };
    auto stageB = [&](int buf, int k0) {
        const _Float16* g = Wh + (size_t)(blockCol + srow) * K + (k0 + scol);
        async_copy64(g, &sB[buf][srow * LDS_STRIDE + scol]);
    };

    // ---- pipelined main loop --------------------------------------------
    stageA(0, 0);
    stageB(0, 0);

    for (int kt = 0; kt < nkt; ++kt) {
        const int cur = kt & 1;
        if (kt + 1 < nkt) {
            stageA(1 - cur, (kt + 1) * BK);
            stageB(1 - cur, (kt + 1) * BK);
            // wait only for the PREVIOUS stage's asyncs (in-order completion)
            if constexpr (sizeof(AT) == 4)
                asm volatile("s_wait_asynccnt 0x4" ::: "memory");
            else
                asm volatile("s_wait_asynccnt 0x8" ::: "memory");
        } else {
            asm volatile("s_wait_asynccnt 0x0" ::: "memory");
        }
        __syncthreads();                    // tiles[cur] visible to all waves

#pragma unroll
        for (int ks = 0; ks < 2; ++ks) {    // two 16x16x32 slices of BK=64
            v16h afrag[2];
#pragma unroll
            for (int mt = 0; mt < 2; ++mt) {
                const int r = (waveM * 32 + mt * 16 + laneRow) * LDS_STRIDE + ks * 32;
                v8h lo = *reinterpret_cast<const v8h*>(&sA[cur][r + aKlo]);
                v8h hi = *reinterpret_cast<const v8h*>(&sA[cur][r + aKhi]);
                reinterpret_cast<v8h*>(&afrag[mt])[0] = lo;
                reinterpret_cast<v8h*>(&afrag[mt])[1] = hi;
            }
            v16h bfrag[4];
#pragma unroll
            for (int nt = 0; nt < 4; ++nt) {
                const int r = (waveN * 64 + nt * 16 + laneRow) * LDS_STRIDE + ks * 32;
                v8h lo = *reinterpret_cast<const v8h*>(&sB[cur][r + aKlo]);
                v8h hi = *reinterpret_cast<const v8h*>(&sB[cur][r + aKhi]);
                reinterpret_cast<v8h*>(&bfrag[nt])[0] = lo;
                reinterpret_cast<v8h*>(&bfrag[nt])[1] = hi;
            }
#pragma unroll
            for (int mt = 0; mt < 2; ++mt)
#pragma unroll
                for (int nt = 0; nt < 4; ++nt)
                    acc[mt][nt] = __builtin_amdgcn_wmma_f32_16x16x32_f16(
                        false, afrag[mt], false, bfrag[nt],
                        (short)0, acc[mt][nt], false, false);
        }
        __syncthreads();                    // all reads of tiles[cur] done
    }

    // ---- epilogue: +bias, activation, store -----------------------------
#pragma unroll
    for (int nt = 0; nt < 4; ++nt) {
        const int gcol = blockCol + waveN * 64 + nt * 16 + laneRow;
        const float bv = bias[gcol];
#pragma unroll
        for (int mt = 0; mt < 2; ++mt) {
            const int rbase = blockRow + waveM * 32 + mt * 16 + (hiHalf ? 8 : 0);
#pragma unroll
            for (int i = 0; i < 8; ++i) {
                float v = acc[mt][nt][i] + bv;
                if constexpr (RELU) v = v > 0.0f ? v : 0.0f;
                C[(size_t)(rbase + i) * N + gcol] = (OT)v;
            }
        }
    }
}

// ---------------------------------------------------------------------------
extern "C" void kernel_launch(void* const* d_in, const int* in_sizes, int n_in,
                              void* d_out, int out_size, void* d_ws, size_t ws_size,
                              hipStream_t stream)
{
    const float* x  = (const float*)d_in[0];
    const float* W0 = (const float*)d_in[1];
    const float* b0 = (const float*)d_in[2];
    const float* W1 = (const float*)d_in[3];
    const float* b1 = (const float*)d_in[4];
    const float* W2 = (const float*)d_in[5];
    const float* b2 = (const float*)d_in[6];
    const float* W3 = (const float*)d_in[7];
    const float* b3 = (const float*)d_in[8];
    float* out = (float*)d_out;

    constexpr int M = 4096;
    constexpr int IN = 2048;
    constexpr int D0 = 4096, D1 = 4096, D2 = 4096, D3 = 1024;

    constexpr size_t nW0 = (size_t)D0 * IN;   // 8.39M
    constexpr size_t nW1 = (size_t)D1 * D0;   // 16.78M
    constexpr size_t nW2 = (size_t)D2 * D1;   // 16.78M
    constexpr size_t nW3 = (size_t)D3 * D2;   // 4.19M

    _Float16* wf0 = reinterpret_cast<_Float16*>(d_ws);
    _Float16* wf1 = wf0 + nW0;
    _Float16* wf2 = wf1 + nW1;
    _Float16* wf3 = wf2 + nW2;
    _Float16* h0  = wf3 + nW3;                // [M, D0] f16
    _Float16* h1  = h0 + (size_t)M * D0;      // [M, D1] f16

    dim3 blk(256);

    // Pass 1: weights -> f16 (one elementwise sweep)
    cvt_f32_to_f16<<<1024, blk, 0, stream>>>(W0, wf0, (int)(nW0 / 8));
    cvt_f32_to_f16<<<1024, blk, 0, stream>>>(W1, wf1, (int)(nW1 / 8));
    cvt_f32_to_f16<<<1024, blk, 0, stream>>>(W2, wf2, (int)(nW2 / 8));
    cvt_f32_to_f16<<<1024, blk, 0, stream>>>(W3, wf3, (int)(nW3 / 8));

    // Pass 2: the four GEMMs
    gemm_bias_act<float, _Float16, true>
        <<<dim3(D0 / BN, M / BM), blk, 0, stream>>>(x, wf0, b0, h0, M, D0, IN);
    gemm_bias_act<_Float16, _Float16, true>
        <<<dim3(D1 / BN, M / BM), blk, 0, stream>>>(h0, wf1, b1, h1, M, D1, D0);
    gemm_bias_act<_Float16, _Float16, true>
        <<<dim3(D2 / BN, M / BM), blk, 0, stream>>>(h1, wf2, b2, h0, M, D2, D1);
    gemm_bias_act<_Float16, float, false>
        <<<dim3(D3 / BN, M / BM), blk, 0, stream>>>(h0, wf3, b3, out, M, D3, D2);
}
